// RN_20023137534695
// MI455X (gfx1250) — matrix operand order repaired and checked by reading, not compiled, not run on hardware
//
#include <hip/hip_runtime.h>

#define DI __device__ __forceinline__

typedef __attribute__((ext_vector_type(16))) __bf16 v16bf;
typedef __attribute__((ext_vector_type(8)))  float  v8f;

union Frag16 { v16bf v; unsigned int d[8]; };

// native hardware bf16 conversions (RNE), avoids VALU emulation
DI unsigned short f2bf(float x){
  return __builtin_bit_cast(unsigned short, (__bf16)x);
}
DI float bf2f(unsigned short h){
  return (float)__builtin_bit_cast(__bf16, h);
}

// A-matrix 16(M)x32(K) bf16 k-index for dword p, lane-half h  (ISA 7.12.2)
DI int frag_kA(int p, int half){ return ((p & 3) * 2) + ((p >> 2) * 16) + half * 8; }
// B-matrix 32(K)x16(N) bf16 k-index, same k order as A
DI int frag_kB(int p, int half){ return ((p & 3) * 2) + ((p >> 2) * 16) + half * 8; }

// A fragment from a bf16 buffer (LDS), row stride ld elements. (merges to 2x ds_load_b128)
DI v16bf load_fragA(const unsigned short* base, int ld, int m0, int k0, int lane){
  int idx = lane & 15, half = lane >> 4;
  Frag16 f;
  const unsigned short* row = base + (size_t)(m0 + idx) * ld + k0;
  #pragma unroll
  for (int p = 0; p < 8; ++p)
    f.d[p] = *(const unsigned int*)(row + frag_kA(p, half));
  return f.v;
}

// B fragment from transposed weights Wt[N][Kpad] (k contiguous), global. (2x global_load_b128)
DI v16bf load_fragB(const unsigned short* Wt, int kpad, int n0, int k0, int lane){
  int idx = lane & 15, half = lane >> 4;
  Frag16 f;
  const unsigned short* row = Wt + (size_t)(n0 + idx) * kpad + k0;
  #pragma unroll
  for (int p = 0; p < 8; ++p)
    f.d[p] = *(const unsigned int*)(row + frag_kB(p, half));
  return f.v;
}

DI v8f wmma_bf16(v16bf a, v16bf b, v8f c){
  return __builtin_amdgcn_wmma_f32_16x16x32_bf16(false, a, false, b, (short)0, c, false, false);
}

DI v8f zero8(){ v8f z = {0.f,0.f,0.f,0.f,0.f,0.f,0.f,0.f}; return z; }

// store D tile (+bias, optional relu) as bf16 into a buffer with row stride ld
DI void store_tile(unsigned short* out, int ld, int m0, int n0, v8f c,
                   float bias, bool dorelu, int lane){
  int n = n0 + (lane & 15), half = lane >> 4;
  #pragma unroll
  for (int v = 0; v < 8; ++v){
    float val = c[v] + bias;
    if (dorelu) val = fmaxf(val, 0.f);
    out[(size_t)(m0 + v + 8 * half) * ld + n] = f2bf(val);
  }
}

// ---------------- workspace layout (bytes) ----------------
static constexpr size_t OFF_WI1T  = 0;                       // [256][32]
static constexpr size_t OFF_WI2T  = OFF_WI1T  + 256*32*2;    // [128][256]
static constexpr size_t OFF_WI3T  = OFF_WI2T  + 128*256*2;   // [112][128]
static constexpr size_t OFF_WG1AT = OFF_WI3T  + 112*128*2;   // [256][128]
static constexpr size_t OFF_WG1BT = OFF_WG1AT + 256*128*2;   // [256][128]
static constexpr size_t OFF_WG2T  = OFF_WG1BT + 256*128*2;   // [256][256]
static constexpr size_t OFF_WG3T  = OFF_WG2T  + 256*256*2;
static constexpr size_t OFF_WG4T  = OFF_WG3T  + 256*256*2;
static constexpr size_t OFF_WFT   = OFF_WG4T  + 256*256*2;
static constexpr size_t OFF_WO1T  = OFF_WFT   + 256*256*2;
static constexpr size_t OFF_WO2T  = OFF_WO1T  + 256*256*2;   // [16][256]
static constexpr size_t OFF_X     = OFF_WO2T  + 16*256*2;    // [98304][128] bf16
static constexpr size_t OFF_G     = OFF_X + (size_t)98304*128*2; // [16384][256] f32

// ---------------- weight prep: transpose + pad + fp32->bf16 ----------------
__global__ void RN_prep(const float* __restrict__ W, unsigned short* __restrict__ Wt,
                        int K, int Nsrc, int Kpad, int Npad, int rowOff, int ldW)
{
  int i = blockIdx.x * blockDim.x + threadIdx.x;
  if (i >= Kpad * Npad) return;
  int n = i / Kpad, k = i % Kpad;
  float v = (k < K && n < Nsrc) ? W[(size_t)(rowOff + k) * ldW + n] : 0.f;
  Wt[(size_t)n * Kpad + k] = f2bf(v);
}

// ---------------- object encoder: 10 -> 256 -> 128 -> 100 ----------------
__global__ __launch_bounds__(256) void RN_encoder(
    const float* __restrict__ tabs,
    const unsigned short* __restrict__ Wi1t, const float* __restrict__ bi1,
    const unsigned short* __restrict__ Wi2t, const float* __restrict__ bi2,
    const unsigned short* __restrict__ Wi3t, const float* __restrict__ bi3,
    unsigned short* __restrict__ xg)
{
  __shared__ unsigned short a1[64 * 40];
  __shared__ unsigned short x1[64 * 264];
  __shared__ unsigned short x2[64 * 136];
  const int tid = threadIdx.x, lane = tid & 31, wid = tid >> 5;
  const int row0 = blockIdx.x * 64;

  for (int e = tid; e < 64 * 32; e += 256){
    int r = e >> 5, k = e & 31;
    float v = (k < 10) ? tabs[(size_t)(row0 + r) * 10 + k] : 0.f;
    a1[r * 40 + k] = f2bf(v);
  }
  __syncthreads();

  // layer i1: [64x32pad] @ [32x256]
  for (int t = wid; t < 64; t += 8){
    int mt = t >> 4, nt = t & 15;
    v8f c = zero8();
    c = wmma_bf16(load_fragA(a1, 40, mt * 16, 0, lane),
                  load_fragB(Wi1t, 32, nt * 16, 0, lane), c);
    store_tile(x1, 264, mt * 16, nt * 16, c, bi1[nt * 16 + (lane & 15)], true, lane);
  }
  __syncthreads();

  // layer i2: [64x256] @ [256x128]
  for (int t = wid; t < 32; t += 8){
    int mt = t >> 3, nt = t & 7;
    v8f c = zero8();
    #pragma unroll
    for (int ks = 0; ks < 8; ++ks)
      c = wmma_bf16(load_fragA(x1, 264, mt * 16, ks * 32, lane),
                    load_fragB(Wi2t, 256, nt * 16, ks * 32, lane), c);
    store_tile(x2, 136, mt * 16, nt * 16, c, bi2[nt * 16 + (lane & 15)], true, lane);
  }
  __syncthreads();

  // layer i3: [64x128] @ [128x112pad], no relu, write x (128-wide rows) to global
  for (int t = wid; t < 28; t += 8){
    int mt = t / 7, nt = t % 7;
    v8f c = zero8();
    #pragma unroll
    for (int ks = 0; ks < 4; ++ks)
      c = wmma_bf16(load_fragA(x2, 136, mt * 16, ks * 32, lane),
                    load_fragB(Wi3t, 128, nt * 16, ks * 32, lane), c);
    int n = nt * 16 + (lane & 15);
    float bias = (n < 100) ? bi3[n] : 0.f;
    int half = lane >> 4;
    #pragma unroll
    for (int v = 0; v < 8; ++v){
      int m = mt * 16 + v + 8 * half;
      xg[(size_t)(row0 + m) * 128 + n] = f2bf(c[v] + bias);
    }
  }
  for (int e = tid; e < 64 * 16; e += 256){   // zero pad cols 112..127
    int r = e >> 4, k = 112 + (e & 15);
    xg[(size_t)(row0 + r) * 128 + k] = 0;
  }
}

// ---------------- relation g-MLP (dominant kernel) ----------------
DI void glayer(const unsigned short* hin, unsigned short* hout,
               const unsigned short* __restrict__ Wt, const float* __restrict__ bias,
               int wid, int lane)
{
  for (int nti = 0; nti < 2; ++nti){
    int nt = wid * 2 + nti;
    // warm this wave's 8KB weight column (global_prefetch_b8, one line per N-row)
    __builtin_prefetch(Wt + (size_t)(nt * 16 + (lane & 15)) * 256, 0, 3);
    v16bf bw[8];                       // full K=256 B column held in registers
    #pragma unroll
    for (int ks = 0; ks < 8; ++ks) bw[ks] = load_fragB(Wt, 256, nt * 16, ks * 32, lane);
    float bs = bias[nt * 16 + (lane & 15)];
    // 9 m-tiles in 3 groups of 3: three independent accumulation chains in flight
    for (int mtb = 0; mtb < 9; mtb += 3){
      v8f c0 = zero8(), c1 = zero8(), c2 = zero8();
      #pragma unroll
      for (int ks = 0; ks < 8; ++ks){
        v16bf a0 = load_fragA(hin, 264, (mtb + 0) * 16, ks * 32, lane);
        v16bf a1 = load_fragA(hin, 264, (mtb + 1) * 16, ks * 32, lane);
        v16bf a2 = load_fragA(hin, 264, (mtb + 2) * 16, ks * 32, lane);
        c0 = wmma_bf16(a0, bw[ks], c0);
        c1 = wmma_bf16(a1, bw[ks], c1);
        c2 = wmma_bf16(a2, bw[ks], c2);
      }
      store_tile(hout, 264, (mtb + 0) * 16, nt * 16, c0, bs, true, lane);
      store_tile(hout, 264, (mtb + 1) * 16, nt * 16, c1, bs, true, lane);
      store_tile(hout, 264, (mtb + 2) * 16, nt * 16, c2, bs, true, lane);
    }
  }
  __syncthreads();
}

__global__ __launch_bounds__(256) void RN_relation(
    const unsigned short* __restrict__ xg, const float* __restrict__ qst,
    const float* __restrict__ Wg1, const float* __restrict__ bg1,
    const unsigned short* __restrict__ Wg1at, const unsigned short* __restrict__ Wg1bt,
    const unsigned short* __restrict__ Wg2t, const float* __restrict__ bg2,
    const unsigned short* __restrict__ Wg3t, const float* __restrict__ bg3,
    const unsigned short* __restrict__ Wg4t, const float* __restrict__ bg4,
    float* __restrict__ gbuf)
{
  __shared__ __align__(16) unsigned char smem[152064];
  unsigned short* hA = (unsigned short*)smem;                    // 144 x 264 bf16
  unsigned short* hB = (unsigned short*)(smem + 76032);          // 144 x 264 bf16
  unsigned short* xl = hB;                                       // 32 x 136 (alias, dead before h2)
  unsigned short* Pl = (unsigned short*)(smem + 76032 + 8704);   // 32 x 264 (alias)
  unsigned short* Ql = (unsigned short*)(smem + 76032 + 25600);  // 32 x 264 (alias)
  float*          rl = (float*)(smem + 76032 + 42496);           // 4 x 256  (alias)

  const int tid = threadIdx.x, lane = tid & 31, wid = tid >> 5;
  const int b0 = blockIdx.x * 4;                                 // 4 batches per WG

  // stage x rows (24 live + 8 zero-pad) into LDS
  for (int e = tid; e < 32 * 136; e += 256){
    int r = e / 136, k = e % 136;
    unsigned short v = 0;
    if (r < 24 && k < 128) v = xg[(size_t)(b0 * 6 + r) * 128 + k];
    xl[e] = v;
  }
  // r = qst @ Wg1[200:211]   (tiny, fp32 VALU)
  {
    int n = tid;
    for (int bb = 0; bb < 4; ++bb){
      float acc = 0.f;
      #pragma unroll
      for (int q = 0; q < 11; ++q)
        acc += qst[(size_t)(b0 + bb) * 11 + q] * Wg1[(size_t)(200 + q) * 256 + n];
      rl[bb * 256 + n] = acc;
    }
  }
  __syncthreads();

  // P = x @ Wg1[0:100], Q = x @ Wg1[100:200]   (M=32pad, K=128pad, N=256)
  for (int t = wid; t < 64; t += 8){
    int mat = t >> 5, mt = (t >> 4) & 1, nt = t & 15;
    const unsigned short* Wt = mat ? Wg1bt : Wg1at;
    v8f c = zero8();
    #pragma unroll
    for (int ks = 0; ks < 4; ++ks)
      c = wmma_bf16(load_fragA(xl, 136, mt * 16, ks * 32, lane),
                    load_fragB(Wt, 128, nt * 16, ks * 32, lane), c);
    store_tile(mat ? Ql : Pl, 264, mt * 16, nt * 16, c, 0.f, false, lane);
  }
  __syncthreads();

  // h1[b,a,c] = relu(P[b,c] + Q[b,a] + r[b] + bg1)  -> hA (144 rows)
  for (int e = tid; e < 144 * 256; e += 256){
    int row = e >> 8, n = e & 255;
    int bb = row / 36, p = row % 36;
    int ai = p / 6, ci = p % 6;
    float v = bf2f(Pl[(bb * 6 + ci) * 264 + n]) + bf2f(Ql[(bb * 6 + ai) * 264 + n])
            + rl[bb * 256 + n] + bg1[n];
    hA[row * 264 + n] = f2bf(fmaxf(v, 0.f));
  }
  __syncthreads();

  glayer(hA, hB, Wg2t, bg2, wid, lane);   // h2
  glayer(hB, hA, Wg3t, bg3, wid, lane);   // h3
  glayer(hA, hB, Wg4t, bg4, wid, lane);   // h4

  // sum-pool over 36 pairs -> g
  {
    int n = tid;
    for (int bb = 0; bb < 4; ++bb){
      float s = 0.f;
      #pragma unroll 6
      for (int p = 0; p < 36; ++p) s += bf2f(hB[(bb * 36 + p) * 264 + n]);
      gbuf[(size_t)(b0 + bb) * 256 + n] = s;
    }
  }
}

// ---------------- head: f -> o -> logits -> log_softmax ----------------
__global__ __launch_bounds__(256) void RN_head(
    const float* __restrict__ gbuf,
    const unsigned short* __restrict__ Wft,  const float* __restrict__ bfv,
    const unsigned short* __restrict__ Wo1t, const float* __restrict__ bo1,
    const unsigned short* __restrict__ Wo2t, const float* __restrict__ bo2,
    float* __restrict__ out)
{
  __shared__ unsigned short gl[64 * 264];
  __shared__ unsigned short fl[64 * 264];
  __shared__ unsigned short ol[64 * 264];
  __shared__ float lg[64 * 16];
  const int tid = threadIdx.x, lane = tid & 31, wid = tid >> 5;
  const int row0 = blockIdx.x * 64;

  for (int e = tid; e < 64 * 264; e += 256){
    int r = e / 264, k = e % 264;
    float v = (k < 256) ? gbuf[(size_t)(row0 + r) * 256 + k] : 0.f;
    gl[e] = f2bf(v);
  }
  __syncthreads();

  for (int t = wid; t < 64; t += 8){
    int mt = t >> 4, nt = t & 15;
    v8f c = zero8();
    #pragma unroll
    for (int ks = 0; ks < 8; ++ks)
      c = wmma_bf16(load_fragA(gl, 264, mt * 16, ks * 32, lane),
                    load_fragB(Wft, 256, nt * 16, ks * 32, lane), c);
    store_tile(fl, 264, mt * 16, nt * 16, c, bfv[nt * 16 + (lane & 15)], true, lane);
  }
  __syncthreads();

  for (int t = wid; t < 64; t += 8){
    int mt = t >> 4, nt = t & 15;
    v8f c = zero8();
    #pragma unroll
    for (int ks = 0; ks < 8; ++ks)
      c = wmma_bf16(load_fragA(fl, 264, mt * 16, ks * 32, lane),
                    load_fragB(Wo1t, 256, nt * 16, ks * 32, lane), c);
    store_tile(ol, 264, mt * 16, nt * 16, c, bo1[nt * 16 + (lane & 15)], true, lane);
  }
  __syncthreads();

  if (wid < 4){
    int mt = wid;
    v8f c = zero8();
    #pragma unroll
    for (int ks = 0; ks < 8; ++ks)
      c = wmma_bf16(load_fragA(ol, 264, mt * 16, ks * 32, lane),
                    load_fragB(Wo2t, 256, 0, ks * 32, lane), c);
    int n = lane & 15, half = lane >> 4;
    float bias = (n < 10) ? bo2[n] : 0.f;
    #pragma unroll
    for (int v = 0; v < 8; ++v)
      lg[(mt * 16 + v + 8 * half) * 16 + n] = c[v] + bias;
  }
  __syncthreads();

  if (tid < 64){
    int r = tid;
    float mx = lg[r * 16];
    #pragma unroll
    for (int j = 1; j < 10; ++j) mx = fmaxf(mx, lg[r * 16 + j]);
    float s = 0.f;
    #pragma unroll
    for (int j = 0; j < 10; ++j) s += expf(lg[r * 16 + j] - mx);
    float lse = logf(s);
    #pragma unroll
    for (int j = 0; j < 10; ++j)
      out[(size_t)(row0 + r) * 10 + j] = lg[r * 16 + j] - mx - lse;
  }
}

// ---------------- launch ----------------
extern "C" void kernel_launch(void* const* d_in, const int* in_sizes, int n_in,
                              void* d_out, int out_size, void* d_ws, size_t ws_size,
                              hipStream_t stream) {
  (void)in_sizes; (void)n_in; (void)out_size; (void)ws_size;
  const float* tabs = (const float*)d_in[0];
  const float* qst  = (const float*)d_in[1];
  const float* Wi1  = (const float*)d_in[2];  const float* bi1 = (const float*)d_in[3];
  const float* Wi2  = (const float*)d_in[4];  const float* bi2 = (const float*)d_in[5];
  const float* Wi3  = (const float*)d_in[6];  const float* bi3 = (const float*)d_in[7];
  const float* Wg1  = (const float*)d_in[8];  const float* bg1 = (const float*)d_in[9];
  const float* Wg2  = (const float*)d_in[10]; const float* bg2 = (const float*)d_in[11];
  const float* Wg3  = (const float*)d_in[12]; const float* bg3 = (const float*)d_in[13];
  const float* Wg4  = (const float*)d_in[14]; const float* bg4 = (const float*)d_in[15];
  const float* Wf   = (const float*)d_in[16]; const float* bfv = (const float*)d_in[17];
  const float* Wo1  = (const float*)d_in[18]; const float* bo1 = (const float*)d_in[19];
  const float* Wo2  = (const float*)d_in[20]; const float* bo2 = (const float*)d_in[21];

  unsigned char* ws = (unsigned char*)d_ws;
  unsigned short* Wi1t  = (unsigned short*)(ws + OFF_WI1T);
  unsigned short* Wi2t  = (unsigned short*)(ws + OFF_WI2T);
  unsigned short* Wi3t  = (unsigned short*)(ws + OFF_WI3T);
  unsigned short* Wg1at = (unsigned short*)(ws + OFF_WG1AT);
  unsigned short* Wg1bt = (unsigned short*)(ws + OFF_WG1BT);
  unsigned short* Wg2t  = (unsigned short*)(ws + OFF_WG2T);
  unsigned short* Wg3t  = (unsigned short*)(ws + OFF_WG3T);
  unsigned short* Wg4t  = (unsigned short*)(ws + OFF_WG4T);
  unsigned short* Wft   = (unsigned short*)(ws + OFF_WFT);
  unsigned short* Wo1t  = (unsigned short*)(ws + OFF_WO1T);
  unsigned short* Wo2t  = (unsigned short*)(ws + OFF_WO2T);
  unsigned short* xg    = (unsigned short*)(ws + OFF_X);
  float*          gbuf  = (float*)(ws + OFF_G);

  auto prep = [&](const float* W, unsigned short* Wt, int K, int Nsrc,
                  int Kpad, int Npad, int rowOff, int ldW){
    int total = Kpad * Npad;
    RN_prep<<<(total + 255) / 256, 256, 0, stream>>>(W, Wt, K, Nsrc, Kpad, Npad, rowOff, ldW);
  };
  prep(Wi1, Wi1t,  10, 256,  32, 256,   0, 256);
  prep(Wi2, Wi2t, 256, 128, 256, 128,   0, 128);
  prep(Wi3, Wi3t, 128, 100, 128, 112,   0, 100);
  prep(Wg1, Wg1at,100, 256, 128, 256,   0, 256);
  prep(Wg1, Wg1bt,100, 256, 128, 256, 100, 256);
  prep(Wg2, Wg2t, 256, 256, 256, 256,   0, 256);
  prep(Wg3, Wg3t, 256, 256, 256, 256,   0, 256);
  prep(Wg4, Wg4t, 256, 256, 256, 256,   0, 256);
  prep(Wf,  Wft,  256, 256, 256, 256,   0, 256);
  prep(Wo1, Wo1t, 256, 256, 256, 256,   0, 256);
  prep(Wo2, Wo2t, 256,  10, 256,  16,   0,  10);

  RN_encoder<<<1536, 256, 0, stream>>>(tabs, Wi1t, bi1, Wi2t, bi2, Wi3t, bi3, xg);
  RN_relation<<<4096, 256, 0, stream>>>(xg, qst, Wg1, bg1, Wg1at, Wg1bt,
                                        Wg2t, bg2, Wg3t, bg3, Wg4t, bg4, gbuf);
  RN_head<<<256, 256, 0, stream>>>(gbuf, Wft, bfv, Wo1t, bo1, Wo2t, bo2, (float*)d_out);
}
